// SelfInteraction_29738353557734
// MI455X (gfx1250) — matrix operand order repaired and checked by reading, not compiled, not run on hardware
//
#include <hip/hip_runtime.h>
#include <math.h>

#define CC 128
#define DD 128
#define NBATCH 8192
#define XW 512          // C + 3C
#define NKT 512         // C*C / 32 K-tiles per pass
#define WPB 4           // waves per block

typedef __attribute__((ext_vector_type(16))) _Float16 v16h;
typedef __attribute__((ext_vector_type(8)))  _Float16 v8h;
typedef __attribute__((ext_vector_type(8)))  float    v8f;

union V16U { v16h v; v8h h[2]; };

static __device__ __forceinline__ v8h splat8(_Float16 s) {
  v8h r = {s, s, s, s, s, s, s, s};
  return r;
}

// ---------------------------------------------------------------------------
// Kernel 1: repack f32 weights W[u][v][w] into f16 WMMA-B tiles in d_ws.
// B tile for (kt, wt): per lane 16 contiguous halves:
//   lane = (kk>=16)*16 + (w&15),  halves h=0..15 hold K = kt*32 + (lane>=16)*16 + h
// so the GEMM kernel fetches each lane's B fragment with two b128 loads.
// ---------------------------------------------------------------------------
__global__ void pack_weights_kernel(const float* __restrict__ Wss,
                                    const float* __restrict__ Wvv,
                                    const float* __restrict__ Wsv,
                                    const float* __restrict__ Wvs,
                                    _Float16* __restrict__ dst) {
  const float* srcs[4] = {Wss, Wvv, Wsv, Wvs};
  const size_t per = (size_t)CC * CC * DD;        // 2^21
  const size_t total = 4 * per;
  for (size_t i = (size_t)blockIdx.x * blockDim.x + threadIdx.x; i < total;
       i += (size_t)gridDim.x * blockDim.x) {
    int widx = (int)(i >> 21);
    int e    = (int)(i & (per - 1));
    int u   = e >> 14;            // /(C*D)
    int rem = e & 16383;
    int v   = rem >> 7;
    int w   = rem & 127;
    float val = srcs[widx][e];
    int K    = (u << 7) | v;      // pair index, K dimension of the big GEMM
    int kt   = K >> 5;
    int kk   = K & 31;
    int lane = ((kk >> 4) << 4) | (w & 15);
    int h    = kk & 15;
    int wt   = w >> 4;
    size_t d = ((size_t)widx << 21) +
               ((((size_t)kt * 8 + wt) * 32 + (size_t)lane) << 4) + h;
    dst[d] = (_Float16)val;
  }
}

// ---------------------------------------------------------------------------
// Kernel 2: main WMMA GEMM. One wave owns a 16-row z-tile.
// A fragments (16x32 f16) generated in-register from LDS x-tile; B streamed
// from the packed f16 weights (L2 resident). f32 accumulation.
// ---------------------------------------------------------------------------
__global__ __launch_bounds__(WPB * 32) void tp_wmma_kernel(
    const float* __restrict__ x,
    const _Float16* __restrict__ Wp,
    float* __restrict__ out) {
  __shared__ __align__(16) _Float16 sx[WPB * 8192];   // 64 KB: per-wave x tile
  const int lane = threadIdx.x & 31;
  const int wave = threadIdx.x >> 5;
  const int z0 = (blockIdx.x * WPB + wave) * 16;

  _Float16* x0h = &sx[wave * 8192];       // [16][128] row-major f16
  _Float16* x1h = x0h + 16 * CC;          // [3][16][128] SoA over k

  // load + convert this wave's 16 rows of x
  for (int i = lane; i < 16 * XW; i += 32) {
    int z = i >> 9;
    int c = i & 511;
    _Float16 hv = (_Float16)x[(size_t)(z0 + z) * XW + c];
    if (c < CC) {
      x0h[z * CC + c] = hv;
    } else {
      int idx = c - CC;
      int v = idx / 3;
      int k = idx - 3 * v;
      x1h[(k * 16 + z) * CC + v] = hv;
    }
  }
  __syncthreads();

  const int zr = lane & 15;       // A-row / D-col index
  const int hi = lane >> 4;       // lane half selects K sub-run
  const _Float16* r0  = x0h + zr * CC;
  const _Float16* r1a = x1h + zr * CC;
  const _Float16* r1b = x1h + (16 * CC) + zr * CC;
  const _Float16* r1c = x1h + (32 * CC) + zr * CC;

  const v8h* Bbase = (const v8h*)Wp;          // slice stride = 2^18 v8h (4 MB)
  const size_t laneoff = (size_t)lane * 2;
  const v8f vzero = {0.f, 0.f, 0.f, 0.f, 0.f, 0.f, 0.f, 0.f};
  const _Float16 inv3 = (_Float16)0.5773502691896258f;  // 1/sqrt(3)
  const float pw = 0.005524271728019903f;  // 1/sqrt(2*C*C) == pw1*inv_sqrt3

#define WMMA_NTILE_LOOP(BW)                                                    \
  {                                                                            \
    size_t bi = (size_t)kt * 512 + laneoff;                                    \
    _Pragma("unroll")                                                          \
    for (int wt = 0; wt < 8; ++wt) {                                           \
      V16U Bt; Bt.h[0] = (BW)[bi]; Bt.h[1] = (BW)[bi + 1];                     \
      acc[wt] = __builtin_amdgcn_wmma_f32_16x16x32_f16(                        \
          false, A.v, false, Bt.v, (short)0, acc[wt], false, false);           \
      bi += 64;                                                                \
    }                                                                          \
  }

  // ---------------- phase 0: y0 = ss + inv3*vv ----------------
  {
    v8f acc[8];
#pragma unroll
    for (int t = 0; t < 8; ++t) acc[t] = vzero;

    // pass 1: A = x0 (x) x0   vs  W_ss
    {
      const v8h* Bw = Bbase;
      for (int kt = 0; kt < NKT; ++kt) {
        const int u = kt >> 2, vb = (kt & 3) << 5;
        v8h s = splat8(r0[u]);
        V16U A;
        A.h[0] = (*(const v8h*)(r0 + vb + hi * 16)) * s;
        A.h[1] = (*(const v8h*)(r0 + vb + 8 + hi * 16)) * s;
        WMMA_NTILE_LOOP(Bw)
      }
    }
    // pass 2: A = dots (3-term fma), inv_sqrt3 folded into u-side scalars
    {
      const v8h* Bw = Bbase + ((size_t)1 << 18);
      for (int kt = 0; kt < NKT; ++kt) {
        const int u = kt >> 2, vb = (kt & 3) << 5;
        v8h s0 = splat8(r1a[u] * inv3);
        v8h s1 = splat8(r1b[u] * inv3);
        v8h s2 = splat8(r1c[u] * inv3);
        V16U A;
        A.h[0] = (*(const v8h*)(r1a + vb + hi * 16)) * s0 +
                 (*(const v8h*)(r1b + vb + hi * 16)) * s1 +
                 (*(const v8h*)(r1c + vb + hi * 16)) * s2;
        A.h[1] = (*(const v8h*)(r1a + vb + 8 + hi * 16)) * s0 +
                 (*(const v8h*)(r1b + vb + 8 + hi * 16)) * s1 +
                 (*(const v8h*)(r1c + vb + 8 + hi * 16)) * s2;
        WMMA_NTILE_LOOP(Bw)
      }
    }
    // store raw y0*pw (normalized later)
#pragma unroll
    for (int wt = 0; wt < 8; ++wt)
#pragma unroll
      for (int r = 0; r < 8; ++r) {
        int z = z0 + r + 8 * hi;
        int w = wt * 16 + zr;
        out[(size_t)z * XW + w] = acc[wt][r] * pw;
      }
  }

  // ---------------- phases 1..3: yv_k = sv + vs ----------------
  const _Float16* r1k[3] = {r1a, r1b, r1c};
  for (int k = 0; k < 3; ++k) {
    v8f acc[8];
#pragma unroll
    for (int t = 0; t < 8; ++t) acc[t] = vzero;
    const _Float16* rk = r1k[k];

    // pass sv: A = x0[u] * x1k[v]
    {
      const v8h* Bw = Bbase + ((size_t)2 << 18);
      for (int kt = 0; kt < NKT; ++kt) {
        const int u = kt >> 2, vb = (kt & 3) << 5;
        v8h s = splat8(r0[u]);
        V16U A;
        A.h[0] = (*(const v8h*)(rk + vb + hi * 16)) * s;
        A.h[1] = (*(const v8h*)(rk + vb + 8 + hi * 16)) * s;
        WMMA_NTILE_LOOP(Bw)
      }
    }
    // pass vs: A = x1k[u] * x0[v]
    {
      const v8h* Bw = Bbase + ((size_t)3 << 18);
      for (int kt = 0; kt < NKT; ++kt) {
        const int u = kt >> 2, vb = (kt & 3) << 5;
        v8h s = splat8(rk[u]);
        V16U A;
        A.h[0] = (*(const v8h*)(r0 + vb + hi * 16)) * s;
        A.h[1] = (*(const v8h*)(r0 + vb + 8 + hi * 16)) * s;
        WMMA_NTILE_LOOP(Bw)
      }
    }
#pragma unroll
    for (int wt = 0; wt < 8; ++wt)
#pragma unroll
      for (int r = 0; r < 8; ++r) {
        int z = z0 + r + 8 * hi;
        int w = wt * 16 + zr;
        out[(size_t)z * XW + CC + 3 * w + k] = acc[wt][r] * pw;
      }
  }
#undef WMMA_NTILE_LOOP
}

// ---------------------------------------------------------------------------
// Kernel 3: per-row std normalization (ddof=1), in place on d_out.
// ---------------------------------------------------------------------------
__global__ __launch_bounds__(128) void finalize_kernel(float* __restrict__ out) {
  __shared__ float red[128];
  const int z = blockIdx.x;
  const int w = threadIdx.x;
  float* row = out + (size_t)z * XW;
  float y0 = row[w];
  float a = row[CC + 3 * w + 0];
  float b = row[CC + 3 * w + 1];
  float c = row[CC + 3 * w + 2];
  float n1 = sqrtf(a * a + b * b + c * c + 1e-9f);

  float s1, s2, t1, t2;
  red[w] = y0; __syncthreads();
  for (int s = 64; s > 0; s >>= 1) { if (w < s) red[w] += red[w + s]; __syncthreads(); }
  s1 = red[0]; __syncthreads();
  red[w] = y0 * y0; __syncthreads();
  for (int s = 64; s > 0; s >>= 1) { if (w < s) red[w] += red[w + s]; __syncthreads(); }
  s2 = red[0]; __syncthreads();
  red[w] = n1; __syncthreads();
  for (int s = 64; s > 0; s >>= 1) { if (w < s) red[w] += red[w + s]; __syncthreads(); }
  t1 = red[0]; __syncthreads();
  red[w] = n1 * n1; __syncthreads();
  for (int s = 64; s > 0; s >>= 1) { if (w < s) red[w] += red[w + s]; __syncthreads(); }
  t2 = red[0];

  float ms = s1 * (1.f / 128.f);
  float vs = (s2 - 128.f * ms * ms) * (1.f / 127.f);
  float stds = sqrtf(fmaxf(vs, 0.f));
  float mv = t1 * (1.f / 128.f);
  float vvv = (t2 - 128.f * mv * mv) * (1.f / 127.f);
  float stdv = sqrtf(fmaxf(vvv, 0.f));

  row[w] = y0 / (stds + 1e-9f);
  float inv = 1.f / (stdv + 1e-9f);
  row[CC + 3 * w + 0] = a * inv;
  row[CC + 3 * w + 1] = b * inv;
  row[CC + 3 * w + 2] = c * inv;
}

extern "C" void kernel_launch(void* const* d_in, const int* in_sizes, int n_in,
                              void* d_out, int out_size, void* d_ws, size_t ws_size,
                              hipStream_t stream) {
  (void)in_sizes; (void)n_in; (void)out_size; (void)ws_size;
  const float* x   = (const float*)d_in[0];
  const float* Wss = (const float*)d_in[1];
  const float* Wvv = (const float*)d_in[2];
  const float* Wsv = (const float*)d_in[3];
  const float* Wvs = (const float*)d_in[4];
  float* out = (float*)d_out;
  _Float16* Wp = (_Float16*)d_ws;   // needs 4 * 2^21 * 2B = 16 MiB of scratch

  pack_weights_kernel<<<1024, 256, 0, stream>>>(Wss, Wvv, Wsv, Wvs, Wp);
  tp_wmma_kernel<<<NBATCH / 16 / WPB, WPB * 32, 0, stream>>>(x, Wp, out);
  finalize_kernel<<<NBATCH, 128, 0, stream>>>(out);
}
